// DPDG_aligned_69458211111606
// MI455X (gfx1250) — matrix operand order repaired and checked by reading, not compiled, not run on hardware
//
#include <hip/hip_runtime.h>
#include <hip/hip_bf16.h>
#include <math.h>

// ---------------------------------------------------------------------------
// DPDG forward for MI455X (gfx1250, wave32).
// All contractions run through v_wmma_f32_16x16x32_f16 (f32->f16 in-register,
// f32 accumulate). Glue (softmax/ELU/diffusion) is scalar VALU kernels.
// ---------------------------------------------------------------------------

#define Bb   128
#define Nn   128
#define Cc   64
#define Mm   8
#define Hh   128
#define Aa   64
#define Ll   2
#define DFFv 256
#define FFFv 64

#define INV_SQRT_H 0.08838834764831845f   // 1/sqrt(128)

typedef __attribute__((ext_vector_type(16))) _Float16 v16h;
typedef __attribute__((ext_vector_type(8)))  float    v8f;

// Promote a generic pointer to global for codegen: lets InferAddressSpaces
// select global_load/global_store (LOADcnt only) instead of flat (LOAD+DS cnt).
template <typename T>
__device__ __forceinline__ T* gptr(T* p) {
#if defined(__has_builtin)
#if __has_builtin(__builtin_amdgcn_is_shared) && __has_builtin(__builtin_amdgcn_is_private)
    __builtin_assume(!__builtin_amdgcn_is_shared((void*)p));
    __builtin_assume(!__builtin_amdgcn_is_private((void*)p));
#endif
#endif
    return p;
}

__device__ __forceinline__ float eluf(float x) {
    return x > 0.f ? x : (__expf(x) - 1.f);
}
__device__ __forceinline__ float sigm(float x) {
    return 1.f / (1.f + __expf(-x));
}

// ---------------------------------------------------------------------------
// Generic batched strided GEMM: C[bo,bi,i,j] = sum_k A[...]*B[...] (+bias)
// One wave computes a 16 x (NT*16) strip; K consumed in chunks of 32 via
// V_WMMA_F32_16X16X32_F16.  Lane layouts per CDNA5 ISA 7.12.2:
//   A (16x32 f16): lane m=lane&15; elem e -> K = e+(half?8:0) (e<8),
//                  (e-8)+(half?24:16) (e>=8)
//   B (32x16 f16): lane n=lane&15; elem e -> K same as A
//   C/D: row = r+(half?8:0), col = lane&15
// Addressing: uniform scalar base + independent 32-bit per-lane element
// offsets (each advanced by 32*stride per chunk) -> independent global loads.
// GUARD=false assumes Md%16==0, Nd%(NT*16)==0, Kd%32==0 (no bounds checks).
// GUARD=true uses clamped-index loads + selects (no exec branches in loop).
// biasMode: 0 none, 1 bias[j], 2 bias[i].  accum: C += result.
// ---------------------------------------------------------------------------
template<int NT, bool GUARD>
__global__ __launch_bounds__(32)
void k_wmma_gemm(const float* __restrict__ Ap, const float* __restrict__ Bp,
                 float* __restrict__ Cp, const float* __restrict__ biasp,
                 int biasMode, int accum, int Md, int Nd, int Kd, int tilesNB,
                 long sAbo, long sAbi, long sAi, long sAk,
                 long sBbo, long sBbi, long sBk, long sBj,
                 long sCbo, long sCbi, long sCi, long sCj)
{
    const float* A    = gptr(Ap);
    const float* Bmat = gptr(Bp);
    float*       Cm   = gptr(Cp);
    const float* bias = gptr(biasp);

    const int lane = threadIdx.x;          // 0..31
    const int tile = blockIdx.x;
    const int ti   = tile / tilesNB;       // row tile
    const int tjb  = tile % tilesNB;       // column block (NT tiles wide)
    const int half = lane >> 4;
    const int lm   = lane & 15;

    const float* Ab  = A    + (long)blockIdx.z * sAbo + (long)blockIdx.y * sAbi;
    const float* Bb2 = Bmat + (long)blockIdx.z * sBbo + (long)blockIdx.y * sBbi;
    float*       Cb  = Cm   + (long)blockIdx.z * sCbo + (long)blockIdx.y * sCbi;

    const int rowA  = ti * 16 + lm;
    const int iAi   = (int)sAi, iAk = (int)sAk, iBk = (int)sBk, iBj = (int)sBj;

    // per-element K indices (constant across the K loop)
    int akk[16], bkk[16];
#pragma unroll
    for (int e = 0; e < 16; ++e) {
        akk[e] = (e < 8) ? (e + (half ? 8 : 0)) : ((e - 8) + (half ? 24 : 16));
        bkk[e] = (e < 8) ? (e + (half ? 8 : 0)) : ((e - 8) + (half ? 24 : 16));
    }

    v8f acc[NT];
#pragma unroll
    for (int t = 0; t < NT; ++t) { v8f z = {}; acc[t] = z; }

    if (!GUARD) {
        // uniform bases (stay in SGPRs); per-lane 32-bit element offsets.
        const float* bBase[NT];
#pragma unroll
        for (int t = 0; t < NT; ++t)
            bBase[t] = Bb2 + (long)(tjb * (NT * 16) + t * 16) * sBj;   // uniform

        int aOff[16], bOff[16];
#pragma unroll
        for (int e = 0; e < 16; ++e) {
            aOff[e] = rowA * iAi + akk[e] * iAk;
            bOff[e] = lm * iBj + bkk[e] * iBk;
        }
        const int aStep = 32 * iAk, bStep = 32 * iBk;

        for (int k0 = 0; k0 < Kd; k0 += 32) {
            // speculative prefetch of next A chunk (silently dropped if OOB)
            __builtin_prefetch(Ab + aOff[0] + aStep, 0, 1);

            v16h av;
            v16h bv[NT];
#pragma unroll
            for (int e = 0; e < 16; ++e) av[e] = (_Float16)Ab[aOff[e]];
#pragma unroll
            for (int t = 0; t < NT; ++t) {
#pragma unroll
                for (int e = 0; e < 16; ++e) bv[t][e] = (_Float16)bBase[t][bOff[e]];
            }
#pragma unroll
            for (int t = 0; t < NT; ++t)
                acc[t] = __builtin_amdgcn_wmma_f32_16x16x32_f16(
                             false, av, false, bv[t], (short)0, acc[t], false, false);

#pragma unroll
            for (int e = 0; e < 16; ++e) { aOff[e] += aStep; bOff[e] += bStep; }
        }
    } else {
        // guarded path: clamp indices, always load, select with cndmask.
        const bool rOK  = rowA < Md;
        const int  rowC = rOK ? rowA : 0;
        const int  col0 = tjb * 16 + lm;
        const bool cOK  = col0 < Nd;
        const int  colC = cOK ? col0 : 0;

        for (int k0 = 0; k0 < Kd; k0 += 32) {
            v16h av, bv;
#pragma unroll
            for (int e = 0; e < 16; ++e) {
                int ka = k0 + akk[e];
                bool okA = rOK && (ka < Kd);
                float va = Ab[rowC * iAi + (okA ? ka : 0) * iAk];
                av[e] = okA ? (_Float16)va : (_Float16)0.f;

                int kb = k0 + bkk[e];
                bool okB = cOK && (kb < Kd);
                float vb = Bb2[colC * iBj + (okB ? kb : 0) * iBk];
                bv[e] = okB ? (_Float16)vb : (_Float16)0.f;
            }
            acc[0] = __builtin_amdgcn_wmma_f32_16x16x32_f16(
                         false, av, false, bv, (short)0, acc[0], false, false);
        }
    }

    // epilogue
#pragma unroll
    for (int t = 0; t < NT; ++t) {
        const int col = tjb * (NT * 16) + t * 16 + lm;
#pragma unroll
        for (int r = 0; r < 8; ++r) {
            const int row = ti * 16 + r + (half ? 8 : 0);
            if (!GUARD || (row < Md && col < Nd)) {
                float v = acc[t][r];
                if (biasMode == 1)      v += bias[col];
                else if (biasMode == 2) v += bias[row];
                long idx = (long)row * sCi + (long)col * sCj;
                if (accum) v += Cb[idx];
                Cb[idx] = v;
            }
        }
    }
}

// ---------------------------------------------------------------------------
// Elementwise / glue kernels
// ---------------------------------------------------------------------------

// h1[b,h,n] = elu( conv1d(x_in^T, deW1, k=3, pad=1) + deb1 )
__global__ void k_conv_enc(const float* __restrict__ xinp, const float* __restrict__ Wp,
                           const float* __restrict__ bvecp, float* __restrict__ h1p)
{
    const float* xin = gptr(xinp); const float* W = gptr(Wp);
    const float* bvec = gptr(bvecp); float* h1 = gptr(h1p);
    int idx = blockIdx.x * blockDim.x + threadIdx.x;
    if (idx >= Bb * Hh * Nn) return;
    int n = idx % Nn; int t = idx / Nn; int h = t % Hh; int b = t / Hh;
    float acc = bvec[h];
    for (int k = 0; k < 3; ++k) {
        int nn = n + k - 1;
        if (nn < 0 || nn >= Nn) continue;
        const float* xr = xin + ((size_t)b * Nn + nn) * Cc;   // x_in[b,nn,:]
        const float* wr = W + (size_t)h * Cc * 3 + k;         // deW1[h,:,k]
        for (int c = 0; c < Cc; ++c) acc += xr[c] * wr[c * 3];
    }
    h1[idx] = eluf(acc);
}

// h1p[b,h,m] = mean over 16 consecutive n of h1[b,h,n]
__global__ void k_pool(const float* __restrict__ h1p, float* __restrict__ h1pp)
{
    const float* h1 = gptr(h1p); float* hp = gptr(h1pp);
    int idx = blockIdx.x * blockDim.x + threadIdx.x;
    if (idx >= Bb * Hh * Mm) return;
    int m = idx % Mm; int t = idx / Mm; int h = t % Hh; int b = t / Hh;
    const float* r = h1 + ((size_t)b * Hh + h) * Nn + m * (Nn / Mm);
    float s = 0.f;
    for (int q = 0; q < Nn / Mm; ++q) s += r[q];
    hp[idx] = s * (1.f / (Nn / Mm));
}

// x[b,c,m,h] = x0[b,c,h]  (broadcast over m)
__global__ void k_bcast(const float* __restrict__ x0p, float* __restrict__ xp)
{
    const float* x0 = gptr(x0p); float* x = gptr(xp);
    int idx = blockIdx.x * blockDim.x + threadIdx.x;
    if (idx >= Bb * Cc * Mm * Hh) return;
    int h = idx % Hh; int t = idx / Hh; t /= Mm; int c = t % Cc; int b = t / Cc;
    x[idx] = x0[((size_t)b * Cc + c) * Hh + h];
}

// gates m / mT: sigmoid(<[x,occ], g{1,2}W[l]> + g{1,2}b[l]) per (b,c,m)
__global__ void k_gate(const float* __restrict__ xp, const float* __restrict__ occp,
                       const float* __restrict__ g1Wp, const float* __restrict__ g1bp,
                       const float* __restrict__ g2Wp, const float* __restrict__ g2bp,
                       float* __restrict__ mvp, float* __restrict__ mtvp, int l)
{
    const float* x = gptr(xp); const float* occ = gptr(occp);
    const float* g1W = gptr(g1Wp); const float* g1b = gptr(g1bp);
    const float* g2W = gptr(g2Wp); const float* g2b = gptr(g2bp);
    float* mv = gptr(mvp); float* mtv = gptr(mtvp);
    int idx = blockIdx.x * blockDim.x + threadIdx.x;
    if (idx >= Bb * Cc * Mm) return;
    int t = idx / Mm; int c = t % Cc; int b = t / Cc;
    const float* xr = x + (size_t)idx * Hh;                  // x[b,c,m,:]
    const float* w1 = g1W + (size_t)l * (Hh + Aa);
    const float* w2 = g2W + (size_t)l * (Hh + Aa);
    float s1 = g1b[l], s2 = g2b[l];
    for (int h = 0; h < Hh; ++h) { float xv = xr[h]; s1 += xv * w1[h]; s2 += xv * w2[h]; }
    const float* orow = occ + ((size_t)b * Cc + c) * Aa;     // occ[b,c,:]
    for (int a = 0; a < Aa; ++a) { float ov = orow[a]; s1 += ov * w1[Hh + a]; s2 += ov * w2[Hh + a]; }
    mv[idx]  = sigm(s1);
    mtv[idx] = sigm(s2);
}

// softmax over a of elu(gate * t * scale), + occ, store transposed.
// mode 0: out[b,m,c,a] (E);  mode 1: out[b,m,a,c] (ET)
__global__ void k_softmax_e(const float* __restrict__ T1pp, const float* __restrict__ mvp,
                            const float* __restrict__ occp, float* __restrict__ outp,
                            int mode, float scale)
{
    const float* T1p = gptr(T1pp); const float* mv = gptr(mvp);
    const float* occ = gptr(occp); float* out = gptr(outp);
    int idx = blockIdx.x * blockDim.x + threadIdx.x;
    if (idx >= Bb * Cc * Mm) return;
    int m = idx % Mm; int t = idx / Mm; int c = t % Cc; int b = t / Cc;
    const float* tr = T1p + (size_t)idx * Aa;
    float g = mv[idx] * scale;
    float mx = -3.4e38f;
    for (int a = 0; a < Aa; ++a) { float p = eluf(g * tr[a]); mx = fmaxf(mx, p); }
    float s = 0.f;
    for (int a = 0; a < Aa; ++a) s += __expf(eluf(g * tr[a]) - mx);
    float inv = 1.f / s;
    const float* orow = occ + ((size_t)b * Cc + c) * Aa;
    for (int a = 0; a < Aa; ++a) {
        float sm = __expf(eluf(g * tr[a]) - mx) * inv;
        float v = orow[a] + sm;
        if (mode == 0) out[(((size_t)b * Mm + m) * Cc + c) * Aa + a] = v;
        else           out[(((size_t)b * Mm + m) * Aa + a) * Cc + c] = v;
    }
}

// adj row softmax(elu(.)) then 0.9*adj + 0.1*I, in place; row = (b,m,c)
__global__ void k_adj_softmax(float* __restrict__ adjpp)
{
    float* adjp = gptr(adjpp);
    int idx = blockIdx.x * blockDim.x + threadIdx.x;
    if (idx >= Bb * Mm * Cc) return;
    int c = idx % Cc;
    float* row = adjp + (size_t)idx * Cc;
    float mx = -3.4e38f;
    for (int d = 0; d < Cc; ++d) mx = fmaxf(mx, eluf(row[d]));
    float s = 0.f;
    for (int d = 0; d < Cc; ++d) s += __expf(eluf(row[d]) - mx);
    float inv = 1.f / s;
    for (int d = 0; d < Cc; ++d) {
        float sm = __expf(eluf(row[d]) - mx) * inv;
        row[d] = 0.9f * sm + ((d == c) ? 0.1f : 0.f);
    }
}

// out_adj[b,l,c,d] = mean_m adj[b,m,c,d]
__global__ void k_adj_mean(const float* __restrict__ adjpp, float* __restrict__ outpp, int l)
{
    const float* adjp = gptr(adjpp); float* outp = gptr(outpp);
    int idx = blockIdx.x * blockDim.x + threadIdx.x;
    if (idx >= Bb * Cc * Cc) return;
    int d = idx % Cc; int t = idx / Cc; int c = t % Cc; int b = t / Cc;
    float s = 0.f;
    for (int m = 0; m < Mm; ++m)
        s += adjp[(((size_t)b * Mm + m) * Cc + c) * Cc + d];
    outp[(((size_t)b * Ll + l) * Cc + c) * Cc + d] = s * (1.f / Mm);
}

// s[b,c,m] = mean_h x[b,c,m,h]
__global__ void k_smean(const float* __restrict__ xp, float* __restrict__ sbp)
{
    const float* x = gptr(xp); float* sb = gptr(sbp);
    int idx = blockIdx.x * blockDim.x + threadIdx.x;
    if (idx >= Bb * Cc * Mm) return;
    const float* xr = x + (size_t)idx * Hh;
    float s = 0.f;
    for (int h = 0; h < Hh; ++h) s += xr[h];
    sb[idx] = s * (1.f / Hh);
}

// Ls[b,c,m] = deg[b,m,c]*s[b,c,m] - sum_cc adj[b,m,cc,c]*s[b,cc,m]
__global__ void k_lap(const float* __restrict__ adjpp, const float* __restrict__ sbp,
                      float* __restrict__ lsp)
{
    const float* adjp = gptr(adjpp); const float* sb = gptr(sbp); float* ls = gptr(lsp);
    int idx = blockIdx.x * blockDim.x + threadIdx.x;
    if (idx >= Bb * Mm * Cc) return;
    int c = idx % Cc; int t = idx / Cc; int m = t % Mm; int b = t / Mm;
    const float* abase = adjp + (size_t)t * Cc * Cc;   // adj[b,m,:,:]
    const float* arow  = abase + (size_t)c * Cc;
    float deg = 0.f;
    for (int d = 0; d < Cc; ++d) deg += arow[d];
    float As = 0.f;
    for (int cc = 0; cc < Cc; ++cc)
        As += abase[(size_t)cc * Cc + c] * sb[((size_t)b * Cc + cc) * Mm + m];
    ls[((size_t)b * Cc + c) * Mm + m] = deg * sb[((size_t)b * Cc + c) * Mm + m] - As;
}

// r[b,c] = elu(mean_m s * rW1 + rb1) @ rW2 + rb2
__global__ void k_r(const float* __restrict__ sbp, const float* __restrict__ rW1p,
                    const float* __restrict__ rb1p, const float* __restrict__ rW2p,
                    const float* __restrict__ rb2p, float* __restrict__ rbufp, int l)
{
    const float* sb = gptr(sbp); const float* rW1 = gptr(rW1p); const float* rb1 = gptr(rb1p);
    const float* rW2 = gptr(rW2p); const float* rb2 = gptr(rb2p); float* rbuf = gptr(rbufp);
    int idx = blockIdx.x * blockDim.x + threadIdx.x;
    if (idx >= Bb * Cc) return;
    float sbar = 0.f;
    for (int m = 0; m < Mm; ++m) sbar += sb[(size_t)idx * Mm + m];
    sbar *= (1.f / Mm);
    float acc = rb2[l];
    for (int j = 0; j < 16; ++j)
        acc += eluf(sbar * rW1[l * 16 + j] + rb1[l * 16 + j]) * rW2[l * 16 + j];
    rbuf[idx] = acc;
}

// s_new = s - softplus(kappa)*Ls + r + alpha*dyn
__global__ void k_snew(const float* __restrict__ sbp, const float* __restrict__ lsp,
                       const float* __restrict__ rbufp, const float* __restrict__ dynp,
                       const float* __restrict__ kappap, const float* __restrict__ alphap,
                       float* __restrict__ snp, int l)
{
    const float* sb = gptr(sbp); const float* ls = gptr(lsp);
    const float* rbuf = gptr(rbufp); const float* dyn = gptr(dynp);
    const float* kappa = gptr(kappap); const float* alpha = gptr(alphap);
    float* sn = gptr(snp);
    int idx = blockIdx.x * blockDim.x + threadIdx.x;
    if (idx >= Bb * Cc * Mm) return;
    float k = kappa[l];
    float sp = (k > 20.f) ? k : log1pf(__expf(k));
    sn[idx] = sb[idx] - sp * ls[idx] + rbuf[idx / Mm] + alpha[l] * dyn[idx];
}

// c_out[b,c,m,h] = elu( h[b,h,c,m] + dpW[l,h]*s_new[b,c,m] + dpb[l,h] )
__global__ void k_assemble(const float* __restrict__ hbp, const float* __restrict__ snp,
                           const float* __restrict__ dpWp, const float* __restrict__ dpbp,
                           float* __restrict__ xoutp, int l)
{
    const float* hb = gptr(hbp); const float* sn = gptr(snp);
    const float* dpW = gptr(dpWp); const float* dpb = gptr(dpbp);
    float* xout = gptr(xoutp);
    int idx = blockIdx.x * blockDim.x + threadIdx.x;
    if (idx >= Bb * Cc * Mm * Hh) return;
    int h = idx % Hh; int t = idx / Hh; int m = t % Mm; t /= Mm;
    int c = t % Cc; int b = t / Cc;
    float hv = hb[(((size_t)b * Hh + h) * Cc + c) * Mm + m];
    float cr = dpW[l * Hh + h] * sn[((size_t)b * Cc + c) * Mm + m] + dpb[l * Hh + h];
    xout[idx] = eluf(hv + cr);
}

// cm[b,c,h] = mean_m c_out[b,c,m,h]
__global__ void k_cmean(const float* __restrict__ xp, float* __restrict__ cmp)
{
    const float* x = gptr(xp); float* cm = gptr(cmp);
    int idx = blockIdx.x * blockDim.x + threadIdx.x;
    if (idx >= Bb * Cc * Hh) return;
    int h = idx % Hh; int t = idx / Hh; int c = t % Cc; int b = t / Cc;
    float s = 0.f;
    for (int m = 0; m < Mm; ++m)
        s += x[(((size_t)b * Cc + c) * Mm + m) * Hh + h];
    cm[idx] = s * (1.f / Mm);
}

// d[b,c] = sum_f elu(t3[b,c,f]) * decW2[f] + decb2
__global__ void k_dvec(const float* __restrict__ t3p, const float* __restrict__ decW2p,
                       const float* __restrict__ decb2p, float* __restrict__ dvp)
{
    const float* t3 = gptr(t3p); const float* decW2 = gptr(decW2p);
    const float* decb2 = gptr(decb2p); float* dv = gptr(dvp);
    int idx = blockIdx.x * blockDim.x + threadIdx.x;
    if (idx >= Bb * Cc) return;
    const float* r = t3 + (size_t)idx * DFFv;
    float acc = decb2[0];
    for (int f = 0; f < DFFv; ++f) acc += eluf(r[f]) * decW2[f];
    dv[idx] = acc;
}

// cd[b] = |elu(d[b,:] @ cdW1 + cdb1) @ cdW2 + cdb2|
__global__ void k_cd(const float* __restrict__ dvp, const float* __restrict__ cdW1p,
                     const float* __restrict__ cdb1p, const float* __restrict__ cdW2p,
                     const float* __restrict__ cdb2p, float* __restrict__ outpp)
{
    const float* dv = gptr(dvp); const float* cdW1 = gptr(cdW1p);
    const float* cdb1 = gptr(cdb1p); const float* cdW2 = gptr(cdW2p);
    const float* cdb2 = gptr(cdb2p); float* outp = gptr(outpp);
    int b = blockIdx.x * blockDim.x + threadIdx.x;
    if (b >= Bb) return;
    float acc = cdb2[0];
    for (int j = 0; j < FFFv; ++j) {
        float hj = cdb1[j];
        for (int c = 0; c < Cc; ++c) hj += dv[(size_t)b * Cc + c] * cdW1[c * FFFv + j];
        acc += eluf(hj) * cdW2[j];
    }
    outp[(size_t)Bb * Ll * Cc * Cc + b] = fabsf(acc);
}

// ---------------------------------------------------------------------------
// Host side
// ---------------------------------------------------------------------------
static inline void launch_gemm(hipStream_t s,
    const float* A, const float* B, float* C, const float* bias,
    int biasMode, int accum, int Md, int Nd, int Kd, int Bo, int Bi,
    long sAbo, long sAbi, long sAi, long sAk,
    long sBbo, long sBbi, long sBk, long sBj,
    long sCbo, long sCbi, long sCi, long sCj)
{
    const bool fast = (Md % 16 == 0) && (Nd % 64 == 0) && (Kd % 32 == 0);
    if (fast) {
        int tM = Md / 16, tNB = Nd / 64;
        dim3 grid((unsigned)(tM * tNB), (unsigned)Bi, (unsigned)Bo);
        k_wmma_gemm<4, false><<<grid, dim3(32), 0, s>>>(A, B, C, bias, biasMode, accum,
            Md, Nd, Kd, tNB,
            sAbo, sAbi, sAi, sAk, sBbo, sBbi, sBk, sBj, sCbo, sCbi, sCi, sCj);
    } else {
        int tM = (Md + 15) / 16, tNB = (Nd + 15) / 16;
        dim3 grid((unsigned)(tM * tNB), (unsigned)Bi, (unsigned)Bo);
        k_wmma_gemm<1, true><<<grid, dim3(32), 0, s>>>(A, B, C, bias, biasMode, accum,
            Md, Nd, Kd, tNB,
            sAbo, sAbi, sAi, sAk, sBbo, sBbi, sBk, sBj, sCbo, sCbi, sCi, sCj);
    }
}

static inline dim3 g1d(size_t n, int bs) { return dim3((unsigned)((n + bs - 1) / bs)); }

extern "C" void kernel_launch(void* const* d_in, const int* in_sizes, int n_in,
                              void* d_out, int out_size, void* d_ws, size_t ws_size,
                              hipStream_t stream)
{
    (void)in_sizes; (void)n_in; (void)out_size; (void)ws_size;

    const float* x_in  = (const float*)d_in[0];
    const float* OCCi  = (const float*)d_in[1];
    const float* Wph   = (const float*)d_in[2];
    const float* bph   = (const float*)d_in[3];
    const float* Wp    = (const float*)d_in[4];
    const float* bp    = (const float*)d_in[5];
    const float* ll1W  = (const float*)d_in[6];
    const float* ll1b  = (const float*)d_in[7];
    const float* ll2W  = (const float*)d_in[8];
    const float* ll2b  = (const float*)d_in[9];
    const float* g1W   = (const float*)d_in[10];
    const float* g1b   = (const float*)d_in[11];
    const float* g2W   = (const float*)d_in[12];
    const float* g2b   = (const float*)d_in[13];
    const float* deW1  = (const float*)d_in[14];
    const float* deb1  = (const float*)d_in[15];
    const float* deW2  = (const float*)d_in[16];
    const float* deb2  = (const float*)d_in[17];
    const float* gcW   = (const float*)d_in[18];
    const float* gcb   = (const float*)d_in[19];
    const float* kappa = (const float*)d_in[20];
    const float* alpha = (const float*)d_in[21];
    const float* rW1   = (const float*)d_in[22];
    const float* rb1   = (const float*)d_in[23];
    const float* rW2   = (const float*)d_in[24];
    const float* rb2   = (const float*)d_in[25];
    const float* dpW   = (const float*)d_in[26];
    const float* dpb   = (const float*)d_in[27];
    const float* taW   = (const float*)d_in[28];
    const float* tab   = (const float*)d_in[29];
    const float* decW1 = (const float*)d_in[30];
    const float* decb1 = (const float*)d_in[31];
    const float* decW2 = (const float*)d_in[32];
    const float* decb2 = (const float*)d_in[33];
    const float* cdW1  = (const float*)d_in[34];
    const float* cdb1  = (const float*)d_in[35];
    const float* cdW2  = (const float*)d_in[36];
    const float* cdb2  = (const float*)d_in[37];

    float* out_adj = (float*)d_out;

    // workspace carve (floats)
    float* ws = (float*)d_ws;
    size_t off = 0;
    auto carve = [&](size_t n) { float* p = ws + off; off += n; return p; };
    float* XA   = carve((size_t)Bb * Cc * Mm * Hh);   // x ping
    float* XB   = carve((size_t)Bb * Cc * Mm * Hh);   // x pong / c_out
    float* HOP1 = carve((size_t)Bb * Hh * Cc * Mm);
    float* HOP2 = carve((size_t)Bb * Hh * Cc * Mm);
    float* HB   = carve((size_t)Bb * Hh * Cc * Mm);
    float* E    = carve((size_t)Bb * Mm * Cc * Aa);
    float* ET   = carve((size_t)Bb * Mm * Aa * Cc);
    float* ADJ  = carve((size_t)Bb * Mm * Cc * Cc);
    float* T1   = carve((size_t)Bb * Cc * Mm * Aa);
    float* X0   = carve((size_t)Bb * Cc * Hh);
    float* OCCP = carve((size_t)Bb * Cc * Aa);
    float* H1   = carve((size_t)Bb * Hh * Nn);
    float* H1P  = carve((size_t)Bb * Hh * Mm);
    float* DYN  = carve((size_t)Bb * Cc * Mm);
    float* MV   = carve((size_t)Bb * Cc * Mm);
    float* MTV  = carve((size_t)Bb * Cc * Mm);
    float* SB   = carve((size_t)Bb * Cc * Mm);
    float* LS   = carve((size_t)Bb * Cc * Mm);
    float* RB   = carve((size_t)Bb * Cc);
    float* SN   = carve((size_t)Bb * Cc * Mm);
    float* CM   = carve((size_t)Bb * Cc * Hh);
    float* Z    = carve((size_t)Bb * Cc * Hh);
    float* T3   = carve((size_t)Bb * Cc * DFFv);
    float* DV   = carve((size_t)Bb * Cc);

    const int TB = 256;

    // --- x0[b,c,h] = x_in[b,:,c] @ Wph + bph ---------------------------------
    launch_gemm(stream, x_in, Wph, X0, bph, 1, 0, Cc, Hh, Nn, Bb, 1,
                (long)Nn * Cc, 0, 1, Cc,
                0, 0, Hh, 1,
                (long)Cc * Hh, 0, Hh, 1);
    // --- occ[b,c,a] = OCC[b,:,c] @ Wp + bp -----------------------------------
    launch_gemm(stream, OCCi, Wp, OCCP, bp, 1, 0, Cc, Aa, Nn, Bb, 1,
                (long)Nn * Cc, 0, 1, Cc,
                0, 0, Aa, 1,
                (long)Cc * Aa, 0, Aa, 1);
    // --- dynamic encoder -----------------------------------------------------
    k_conv_enc<<<g1d((size_t)Bb * Hh * Nn, TB), TB, 0, stream>>>(x_in, deW1, deb1, H1);
    k_pool<<<g1d((size_t)Bb * Hh * Mm, TB), TB, 0, stream>>>(H1, H1P);
    launch_gemm(stream, deW2, H1P, DYN, deb2, 2, 0, Cc, Mm, Hh, Bb, 1,
                0, 0, Hh, 1,
                (long)Hh * Mm, 0, Mm, 1,
                (long)Cc * Mm, 0, Mm, 1);
    // --- x = broadcast(x0) over M --------------------------------------------
    k_bcast<<<g1d((size_t)Bb * Cc * Mm * Hh, TB), TB, 0, stream>>>(X0, XA);

    float* xcur = XA;
    float* xnxt = XB;

    for (int l = 0; l < Ll; ++l) {
        // gates m / mT
        k_gate<<<g1d((size_t)Bb * Cc * Mm, TB), TB, 0, stream>>>(
            xcur, OCCP, g1W, g1b, g2W, g2b, MV, MTV, l);

        // t1 = x @ ll1W[l] + ll1b[l]  -> E (softmax, +occ, transpose)
        launch_gemm(stream, xcur, ll1W + (size_t)l * Hh * Aa, T1, ll1b + (size_t)l * Aa,
                    1, 0, Cc * Mm, Aa, Hh, Bb, 1,
                    (long)Cc * Mm * Hh, 0, Hh, 1,
                    0, 0, Aa, 1,
                    (long)Cc * Mm * Aa, 0, Aa, 1);
        k_softmax_e<<<g1d((size_t)Bb * Cc * Mm, TB), TB, 0, stream>>>(T1, MV, OCCP, E, 0, 1.0f);

        // t2 = x @ ll2W[l] + ll2b[l]  -> ET
        launch_gemm(stream, xcur, ll2W + (size_t)l * Hh * Aa, T1, ll2b + (size_t)l * Aa,
                    1, 0, Cc * Mm, Aa, Hh, Bb, 1,
                    (long)Cc * Mm * Hh, 0, Hh, 1,
                    0, 0, Aa, 1,
                    (long)Cc * Mm * Aa, 0, Aa, 1);
        k_softmax_e<<<g1d((size_t)Bb * Cc * Mm, TB), TB, 0, stream>>>(T1, MTV, OCCP, ET, 1, INV_SQRT_H);

        // adj_pre[b,m,c,d] = E[b,m,c,:] @ ET[b,m,:,d]
        launch_gemm(stream, E, ET, ADJ, nullptr, 0, 0, Cc, Cc, Aa, Bb, Mm,
                    (long)Mm * Cc * Aa, (long)Cc * Aa, Aa, 1,
                    (long)Mm * Aa * Cc, (long)Aa * Cc, Cc, 1,
                    (long)Mm * Cc * Cc, (long)Cc * Cc, Cc, 1);
        k_adj_softmax<<<g1d((size_t)Bb * Mm * Cc, TB), TB, 0, stream>>>(ADJ);
        k_adj_mean<<<g1d((size_t)Bb * Cc * Cc, TB), TB, 0, stream>>>(ADJ, out_adj, l);

        // graph diffusion hops: x1[b,f,v,m] = sum_n x[b,n,m,f] * adj[b,m,n,v]
        launch_gemm(stream, xcur, ADJ, HOP1, nullptr, 0, 0, Hh, Cc, Cc, Bb, Mm,
                    (long)Cc * Mm * Hh, Hh, 1, (long)Mm * Hh,
                    (long)Mm * Cc * Cc, (long)Cc * Cc, Cc, 1,
                    (long)Hh * Cc * Mm, 1, (long)Cc * Mm, Mm);
        launch_gemm(stream, HOP1, ADJ, HOP2, nullptr, 0, 0, Hh, Cc, Cc, Bb, Mm,
                    (long)Hh * Cc * Mm, 1, (long)Cc * Mm, Mm,
                    (long)Mm * Cc * Cc, (long)Cc * Cc, Cc, 1,
                    (long)Hh * Cc * Mm, 1, (long)Cc * Mm, Mm);

        // h = gcW[l] @ [xg; x1; x2] + gcb[l]   (three accumulating GEMMs)
        const float* gw = gcW + (size_t)l * Hh * (3 * Hh);
        launch_gemm(stream, gw, xcur, HB, gcb + (size_t)l * Hh, 2, 0,
                    Hh, Cc * Mm, Hh, Bb, 1,
                    0, 0, (long)(3 * Hh), 1,
                    (long)Cc * Mm * Hh, 0, 1, Hh,
                    (long)Hh * Cc * Mm, 0, (long)Cc * Mm, 1);
        launch_gemm(stream, gw + Hh, HOP1, HB, nullptr, 0, 1,
                    Hh, Cc * Mm, Hh, Bb, 1,
                    0, 0, (long)(3 * Hh), 1,
                    (long)Hh * Cc * Mm, 0, (long)Cc * Mm, 1,
                    (long)Hh * Cc * Mm, 0, (long)Cc * Mm, 1);
        launch_gemm(stream, gw + 2 * Hh, HOP2, HB, nullptr, 0, 1,
                    Hh, Cc * Mm, Hh, Bb, 1,
                    0, 0, (long)(3 * Hh), 1,
                    (long)Hh * Cc * Mm, 0, (long)Cc * Mm, 1,
                    (long)Hh * Cc * Mm, 0, (long)Cc * Mm, 1);

        // diffusion layer
        k_smean<<<g1d((size_t)Bb * Cc * Mm, TB), TB, 0, stream>>>(xcur, SB);
        k_lap<<<g1d((size_t)Bb * Mm * Cc, TB), TB, 0, stream>>>(ADJ, SB, LS);
        k_r<<<g1d((size_t)Bb * Cc, TB), TB, 0, stream>>>(SB, rW1, rb1, rW2, rb2, RB, l);
        k_snew<<<g1d((size_t)Bb * Cc * Mm, TB), TB, 0, stream>>>(SB, LS, RB, DYN, kappa, alpha, SN, l);
        k_assemble<<<g1d((size_t)Bb * Cc * Mm * Hh, TB), TB, 0, stream>>>(HB, SN, dpW, dpb, xnxt, l);

        float* tmp = xcur; xcur = xnxt; xnxt = tmp;   // next layer reads c_out
    }

    // --- head: z = mean_m(c_out) @ taW + tab ---------------------------------
    k_cmean<<<g1d((size_t)Bb * Cc * Hh, TB), TB, 0, stream>>>(xcur, CM);
    launch_gemm(stream, CM, taW, Z, tab, 1, 0, Cc, Hh, Hh, Bb, 1,
                (long)Cc * Hh, 0, Hh, 1,
                0, 0, Hh, 1,
                (long)Cc * Hh, 0, Hh, 1);
    launch_gemm(stream, Z, decW1, T3, decb1, 1, 0, Cc, DFFv, Hh, Bb, 1,
                (long)Cc * Hh, 0, Hh, 1,
                0, 0, DFFv, 1,
                (long)Cc * DFFv, 0, DFFv, 1);
    k_dvec<<<g1d((size_t)Bb * Cc, TB), TB, 0, stream>>>(T3, decW2, decb2, DV);
    k_cd<<<g1d((size_t)Bb, 128), 128, 0, stream>>>(DV, cdW1, cdb1, cdW2, cdb2, (float*)d_out);
}
